// TemporalSpatialLocalMultiheadAttention_55911884259738
// MI455X (gfx1250) — matrix operand order repaired and checked by reading, not compiled, not run on hardware
//
#include <hip/hip_runtime.h>

// ---------------- problem constants ----------------
constexpr int T_  = 8;      // T1 == T2
constexpr int HH  = 56;
constexpr int WW  = 56;
constexpr int C_  = 512;
constexpr int NH  = 8;      // heads
constexpr int HD  = 64;     // head dim
constexpr int L_  = 392;    // T*49 tokens per window-sequence
constexpr int LP  = 416;    // padded to multiple of 32 (13*32, 26*16)
constexpr int B_  = 128;    // N * 8 * 8 windows
constexpr int MTOK = L_ * B_;  // 50176 total tokens (multiple of 64)

typedef __attribute__((ext_vector_type(16))) __bf16 v16bf;
typedef __attribute__((ext_vector_type(8)))  __bf16 v8bf;
typedef __attribute__((ext_vector_type(8)))  float  v8f;
typedef __attribute__((ext_vector_type(4)))  int    v4i;

// workspace element counts (bf16)
constexpr size_t SZ_W   = (size_t)3 * C_ * C_;       // in_proj weights bf16
constexpr size_t SZ_OW  = (size_t)C_ * C_;           // out_w bf16
constexpr size_t SZ_QKV = (size_t)B_ * NH * LP * HD; // per Q/K/V tensor
constexpr size_t SZ_O   = (size_t)MTOK * C_;         // attn output bf16

// ---------------- async global->LDS helpers (CDNA5 ASYNCcnt path) ----------------
__device__ __forceinline__ void cp_async16(void* lds, const void* g) {
#if __has_builtin(__builtin_amdgcn_global_load_async_to_lds_b128)
    __builtin_amdgcn_global_load_async_to_lds_b128(
        (v4i*)const_cast<void*>(g), (v4i*)lds, 0, 0);
#else
    *(v8bf*)lds = *(const v8bf*)g;
#endif
}
__device__ __forceinline__ void cp_async_wait() {
#if __has_builtin(__builtin_amdgcn_s_wait_asynccnt)
    __builtin_amdgcn_s_wait_asynccnt(0);
#endif
}

// ---------------- kernel 1: weight fp32 -> bf16 ----------------
__global__ __launch_bounds__(256) void cvt_w_kernel(const float* __restrict__ ipw,
                                                    const float* __restrict__ ow,
                                                    __bf16* __restrict__ wdst) {
    size_t i = (size_t)blockIdx.x * 256 + threadIdx.x;
    if (i < SZ_W)              wdst[i] = (__bf16)ipw[i];
    else if (i < SZ_W + SZ_OW) wdst[i] = (__bf16)ow[i - SZ_W];
}

// ---------------- kernel 2: fused gather + pos-add + projection GEMM ----------------
// blockIdx.x: token tile (64 tokens), blockIdx.y: output tile (64 feats), blockIdx.z: 0=q,1=k,2=v
__global__ __launch_bounds__(256) void proj_kernel(const float* __restrict__ query,
                                                   const float* __restrict__ memory,
                                                   const float* __restrict__ pos,
                                                   const __bf16* __restrict__ Wall,
                                                   const float* __restrict__ bias_all,
                                                   __bf16* __restrict__ Qp,
                                                   __bf16* __restrict__ Kp,
                                                   __bf16* __restrict__ Vpt) {
    __shared__ __bf16 Xs[64][32];
    __shared__ __bf16 Ws[64][32];

    const int z    = blockIdx.z;           // scalar: 0=q, 1=k, 2=v
    const int m0   = blockIdx.x * 64;
    const int n0   = blockIdx.y * 64;
    const int tid  = threadIdx.x;
    const int lane = tid & 31;
    const int wv   = tid >> 5;

    const __bf16* Wmat = Wall + (size_t)z * C_ * C_;
    const float*  bias = bias_all + z * C_;

    // per-thread gather row decode (window permute): row = tid/4, 8 consecutive channels
    const int xr  = tid >> 2;          // 0..63
    const int xc0 = (tid & 3) * 8;     // 0,8,16,24
    const int tIdx = m0 + xr;
    const int b  = tIdx / L_;
    const int l  = tIdx - b * L_;
    const int t  = l / 49, r = l - t * 49;
    const int wi = r / 7,  wj = r - wi * 7;
    const int nb = b >> 6, q2 = b & 63;
    const int qh = q2 >> 3, qw = q2 & 7;
    const int h2 = qh * 7 + wi, w2 = qw * 7 + wj;
    const float* src = (z == 0) ? query : memory;
    const size_t srcBase = ((((size_t)nb * T_ + t) * HH + h2) * WW + w2) * C_;
    const size_t posBase = (z == 0) ? (((size_t)(T_ + t) * 49 + r) * C_)
                                    : (((size_t)t * 49 + r) * C_);

    // wave -> 16x16 subtile mapping: 4 m-blocks x 4 n-blocks, 2 n-blocks per wave
    const int mblk  = wv & 3;
    const int nblk0 = (wv >> 2) * 2;
    const int half  = lane >> 4;
    const int mrow  = mblk * 16 + (lane & 15);
    const int nn    = lane & 15;

    v8f c0 = {}; v8f c1 = {};

    for (int k0 = 0; k0 < C_; k0 += 32) {
        __syncthreads();
        // async-stage W tile (bf16, contiguous): one 16B chunk per thread
        cp_async16(&Ws[xr][xc0], Wmat + (size_t)(n0 + xr) * C_ + k0 + xc0);

        // stage X: vectorized fp32 gather + pos-add (uniform branch on z), pack bf16
        {
            const int cc = k0 + xc0;
            float4 x0 = *(const float4*)(src + srcBase + cc);
            float4 x1 = *(const float4*)(src + srcBase + cc + 4);
            if (z != 2) {   // uniform: blockIdx.z-derived
                const float4 p0 = *(const float4*)(pos + posBase + cc);
                const float4 p1 = *(const float4*)(pos + posBase + cc + 4);
                x0.x += p0.x; x0.y += p0.y; x0.z += p0.z; x0.w += p0.w;
                x1.x += p1.x; x1.y += p1.y; x1.z += p1.z; x1.w += p1.w;
            }
            v8bf px;
            px[0] = (__bf16)x0.x; px[1] = (__bf16)x0.y;
            px[2] = (__bf16)x0.z; px[3] = (__bf16)x0.w;
            px[4] = (__bf16)x1.x; px[5] = (__bf16)x1.y;
            px[6] = (__bf16)x1.z; px[7] = (__bf16)x1.w;
            *(v8bf*)&Xs[xr][xc0] = px;
        }
        cp_async_wait();
        __syncthreads();

        // A fragment (16x32 bf16): lane -> M, VGPR v packs K = 16*half + 2v
        v16bf a;
        #pragma unroll
        for (int v = 0; v < 8; ++v) {
            const int kk = 16 * half + 2 * v;
            a[2 * v]     = Xs[mrow][kk];
            a[2 * v + 1] = Xs[mrow][kk + 1];
        }
        // B fragments (32x16): lane%16 -> N, K = 16*half + 2v
        v16bf b0, b1;
        #pragma unroll
        for (int v = 0; v < 8; ++v) {
            const int kk = 16 * half + 2 * v;
            b0[2 * v]     = Ws[nblk0 * 16 + nn][kk];
            b0[2 * v + 1] = Ws[nblk0 * 16 + nn][kk + 1];
            b1[2 * v]     = Ws[nblk0 * 16 + 16 + nn][kk];
            b1[2 * v + 1] = Ws[nblk0 * 16 + 16 + nn][kk + 1];
        }
        c0 = __builtin_amdgcn_wmma_f32_16x16x32_bf16(false, a, false, b0, (short)0, c0, false, false);
        c1 = __builtin_amdgcn_wmma_f32_16x16x32_bf16(false, a, false, b1, (short)0, c1, false, false);
    }

    // store: fold softmax scale 1/8 into q
    const float scale = (z == 0) ? 0.125f : 1.0f;
    #pragma unroll
    for (int v = 0; v < 8; ++v) {
        const int m  = mblk * 16 + 8 * half + v;
        const int tI = m0 + m;
        const int bb = tI / L_, ll = tI - bb * L_;
        #pragma unroll
        for (int which = 0; which < 2; ++which) {
            const int n   = n0 + (nblk0 + which) * 16 + nn;
            float val = (((which == 0) ? c0[v] : c1[v]) + bias[n]) * scale;
            const int hI = n >> 6, dd = n & 63;
            const size_t bh = (size_t)bb * NH + hI;
            const __bf16 bvv = (__bf16)val;
            if (z == 2)      Vpt[(bh * HD + dd) * LP + ll] = bvv;       // V transposed [bh][d][LP]
            else if (z == 0) Qp[(bh * LP + ll) * HD + dd]  = bvv;       // [bh][LP][64]
            else             Kp[(bh * LP + ll) * HD + dd]  = bvv;
        }
    }
}

// ---------------- kernel 3: attention per (bh), 32 query rows per block ----------------
// grid (13, 1024), 256 threads = 8 waves
__global__ __launch_bounds__(256) void attn_kernel(const __bf16* __restrict__ Qp,
                                                   const __bf16* __restrict__ Kp,
                                                   const __bf16* __restrict__ Vpt,
                                                   __bf16* __restrict__ Obf) {
    __shared__ float  Ssc[32][LP];   // 53 KB: scores then exp(scores - max)
    __shared__ __bf16 Qs[32][HD];    // 4 KB
    __shared__ float  pred[32][8];   // softmax partials
    __shared__ float  rinv[32];

    const int bh = blockIdx.y;
    const int b  = bh >> 3, hh = bh & 7;
    const int m0 = blockIdx.x * 32;
    const int tid = threadIdx.x, lane = tid & 31, wv = tid >> 5;
    const int half = lane >> 4, nn = lane & 15;

    const __bf16* Qbase = Qp  + (size_t)bh * LP * HD;
    const __bf16* Kbase = Kp  + (size_t)bh * LP * HD;
    const __bf16* Vbase = Vpt + (size_t)bh * HD * LP;

    // async-load 32x64 Q tile: exactly 256 x 16B
    {
        const int row = tid >> 3, cc = (tid & 7) * 8;
        cp_async16(&Qs[row][cc], Qbase + (size_t)(m0 + row) * HD + cc);
        cp_async_wait();
    }
    __syncthreads();

    const int mblk = wv & 1;
    const int mrow = mblk * 16 + nn;

    // Q A-fragments for d0=0 and d0=32, loaded once
    v16bf aq0, aq1;
    #pragma unroll
    for (int v = 0; v < 8; ++v) {
        const int kk = 16 * half + 2 * v;
        aq0[2 * v]     = Qs[mrow][kk];
        aq0[2 * v + 1] = Qs[mrow][kk + 1];
        aq1[2 * v]     = Qs[mrow][32 + kk];
        aq1[2 * v + 1] = Qs[mrow][32 + kk + 1];
    }

    // scores: 2 m-blocks x 26 n-blocks of 16x16, distributed over 8 waves
    for (int j = 0; j < 7; ++j) {
        const int nblk = (wv >> 1) + 4 * j;     // wave-uniform
        if (nblk >= LP / 16) break;
        v8f c = {};
        v16bf bk;
        // d0 = 0
        #pragma unroll
        for (int v = 0; v < 8; ++v) {
            const int d = 16 * half + 2 * v;
            const __bf16* kp2 = &Kbase[(size_t)(nblk * 16 + nn) * HD + d];
            bk[2 * v] = kp2[0]; bk[2 * v + 1] = kp2[1];
        }
        c = __builtin_amdgcn_wmma_f32_16x16x32_bf16(false, aq0, false, bk, (short)0, c, false, false);
        // d0 = 32
        #pragma unroll
        for (int v = 0; v < 8; ++v) {
            const int d = 32 + 16 * half + 2 * v;
            const __bf16* kp2 = &Kbase[(size_t)(nblk * 16 + nn) * HD + d];
            bk[2 * v] = kp2[0]; bk[2 * v + 1] = kp2[1];
        }
        c = __builtin_amdgcn_wmma_f32_16x16x32_bf16(false, aq1, false, bk, (short)0, c, false, false);
        // write scores with key-padding mask
        #pragma unroll
        for (int v = 0; v < 8; ++v) {
            const int m   = mblk * 16 + 8 * half + v;
            const int key = nblk * 16 + nn;
            float s = c[v];
            if (key >= L_) s = -1e30f;
            Ssc[m][key] = s;
        }
    }
    __syncthreads();

    // row softmax: 8 threads per row (416 = 8*52); unnormalized exp kept in LDS
    {
        const int srow = tid >> 3;     // 0..31
        const int ssub = tid & 7;      // 0..7
        const int kb = ssub * 52;
        float mx = -1e30f;
        for (int k = kb; k < kb + 52; ++k) mx = fmaxf(mx, Ssc[srow][k]);
        pred[srow][ssub] = mx;
        __syncthreads();
        float rmx = pred[srow][0];
        #pragma unroll
        for (int u = 1; u < 8; ++u) rmx = fmaxf(rmx, pred[srow][u]);
        float sm = 0.f;
        for (int k = kb; k < kb + 52; ++k) {
            const float e = __expf(Ssc[srow][k] - rmx);
            Ssc[srow][k] = e;
            sm += e;
        }
        __syncthreads();
        pred[srow][ssub] = sm;
        __syncthreads();
        if (ssub == 0) {
            float s = 0.f;
            #pragma unroll
            for (int u = 0; u < 8; ++u) s += pred[srow][u];
            rinv[srow] = 1.0f / s;
        }
    }
    __syncthreads();

    // AV: out 32x64, wave -> (mblk = wv&1, d-block = wv>>1)
    const int nblkv = wv >> 1;
    v8f co = {};
    for (int k0 = 0; k0 < LP; k0 += 32) {
        if (k0 + 32 < LP)
            __builtin_prefetch(&Vbase[(size_t)(nblkv * 16 + nn) * LP + k0 + 32], 0, 1);
        v16bf ap, bv;
        #pragma unroll
        for (int v = 0; v < 8; ++v) {
            const int kk = k0 + 16 * half + 2 * v;
            ap[2 * v]     = (__bf16)Ssc[mrow][kk];
            ap[2 * v + 1] = (__bf16)Ssc[mrow][kk + 1];
        }
        #pragma unroll
        for (int v = 0; v < 8; ++v) {
            const int key = k0 + 16 * half + 2 * v;
            const __bf16* vp2 = &Vbase[(size_t)(nblkv * 16 + nn) * LP + key];
            bv[2 * v] = vp2[0]; bv[2 * v + 1] = vp2[1];
        }
        co = __builtin_amdgcn_wmma_f32_16x16x32_bf16(false, ap, false, bv, (short)0, co, false, false);
    }
    #pragma unroll
    for (int v = 0; v < 8; ++v) {
        const int m  = mblk * 16 + 8 * half + v;
        const int ll = m0 + m;
        if (ll < L_) {
            const float val = co[v] * rinv[m];
            const int cchan = hh * HD + nblkv * 16 + nn;
            Obf[((size_t)b * L_ + ll) * C_ + cchan] = (__bf16)val;
        }
    }
}

// ---------------- kernel 4: output projection + reverse window permute ----------------
__global__ __launch_bounds__(256) void outproj_kernel(const __bf16* __restrict__ Obf,
                                                      const __bf16* __restrict__ OW,
                                                      const float* __restrict__ out_b,
                                                      float* __restrict__ out) {
    __shared__ __bf16 Xs[64][32];
    __shared__ __bf16 Ws[64][32];

    const int m0 = blockIdx.x * 64;
    const int n0 = blockIdx.y * 64;
    const int tid = threadIdx.x, lane = tid & 31, wv = tid >> 5;
    const int xr = tid >> 2, xc0 = (tid & 3) * 8;
    const int mblk = wv & 3, nblk0 = (wv >> 2) * 2;
    const int half = lane >> 4, nn = lane & 15;
    const int mrow = mblk * 16 + nn;

    v8f c0 = {}; v8f c1 = {};

    for (int k0 = 0; k0 < C_; k0 += 32) {
        __syncthreads();
        // async-stage both bf16 tiles: 2 x 16B per thread
        cp_async16(&Xs[xr][xc0], Obf + (size_t)(m0 + xr) * C_ + k0 + xc0);
        cp_async16(&Ws[xr][xc0], OW  + (size_t)(n0 + xr) * C_ + k0 + xc0);
        cp_async_wait();
        __syncthreads();

        v16bf a;
        #pragma unroll
        for (int v = 0; v < 8; ++v) {
            const int kk = 16 * half + 2 * v;
            a[2 * v]     = Xs[mrow][kk];
            a[2 * v + 1] = Xs[mrow][kk + 1];
        }
        v16bf b0, b1;
        #pragma unroll
        for (int v = 0; v < 8; ++v) {
            const int kk = 16 * half + 2 * v;
            b0[2 * v]     = Ws[nblk0 * 16 + nn][kk];
            b0[2 * v + 1] = Ws[nblk0 * 16 + nn][kk + 1];
            b1[2 * v]     = Ws[nblk0 * 16 + 16 + nn][kk];
            b1[2 * v + 1] = Ws[nblk0 * 16 + 16 + nn][kk + 1];
        }
        c0 = __builtin_amdgcn_wmma_f32_16x16x32_bf16(false, a, false, b0, (short)0, c0, false, false);
        c1 = __builtin_amdgcn_wmma_f32_16x16x32_bf16(false, a, false, b1, (short)0, c1, false, false);
    }

    // store with reverse window permute into (N, T2, 56, 56, 512) fp32
    #pragma unroll
    for (int v = 0; v < 8; ++v) {
        const int m  = mblk * 16 + 8 * half + v;
        const int tI = m0 + m;
        const int bb = tI / L_, ll = tI - bb * L_;
        const int t  = ll / 49, r = ll - t * 49;
        const int wi = r / 7,   wj = r - wi * 7;
        const int nb = bb >> 6, q2 = bb & 63;
        const int qh = q2 >> 3, qw = q2 & 7;
        const int h2 = qh * 7 + wi, w2 = qw * 7 + wj;
        const size_t obase = ((((size_t)nb * T_ + t) * HH + h2) * WW + w2) * C_;
        #pragma unroll
        for (int which = 0; which < 2; ++which) {
            const int n = n0 + (nblk0 + which) * 16 + nn;
            const float val = ((which == 0) ? c0[v] : c1[v]) + out_b[n];
            out[obase + n] = val;
        }
    }
}

extern "C" void kernel_launch(void* const* d_in, const int* in_sizes, int n_in,
                              void* d_out, int out_size, void* d_ws, size_t ws_size,
                              hipStream_t stream) {
    (void)in_sizes; (void)n_in; (void)out_size; (void)ws_size;
    const float* memory = (const float*)d_in[0];
    const float* query  = (const float*)d_in[1];
    const float* pos    = (const float*)d_in[2];
    const float* ipw    = (const float*)d_in[3];
    const float* ipb    = (const float*)d_in[4];
    const float* ow     = (const float*)d_in[5];
    const float* ob     = (const float*)d_in[6];
    float* out = (float*)d_out;

    __bf16* Wall = (__bf16*)d_ws;
    __bf16* OWp  = Wall + SZ_W;
    __bf16* Qp   = OWp + SZ_OW;
    __bf16* Kp   = Qp + SZ_QKV;
    __bf16* Vpt  = Kp + SZ_QKV;
    __bf16* Obf  = Vpt + SZ_QKV;

    const int cvt_blocks = (int)((SZ_W + SZ_OW + 255) / 256);
    cvt_w_kernel<<<cvt_blocks, 256, 0, stream>>>(ipw, ow, Wall);
    proj_kernel<<<dim3(MTOK / 64, C_ / 64, 3), 256, 0, stream>>>(query, memory, pos, Wall, ipb, Qp, Kp, Vpt);
    attn_kernel<<<dim3(LP / 32, B_ * NH), 256, 0, stream>>>(Qp, Kp, Vpt, Obf);
    outproj_kernel<<<dim3(MTOK / 64, C_ / 64), 256, 0, stream>>>(Obf, OWp, ob, out);
}
